// DecoderBlock_68135361184382
// MI455X (gfx1250) — matrix operand order repaired and checked by reading, not compiled, not run on hardware
//
#include <hip/hip_runtime.h>
#include <math.h>

typedef _Float16 half_t;
typedef __attribute__((ext_vector_type(16))) _Float16 v16h;
typedef __attribute__((ext_vector_type(8)))  _Float16 v8h;
typedef __attribute__((ext_vector_type(8)))  float    v8f;

#define S_   1024
#define DIM_ 1024
#define NQ_  16
#define NKV_ 4
#define HD_  64
#define HID_ 2048
#define E_   8

// ---------------------------------------------------------------------------
// elementwise: f32 -> f16 convert
// ---------------------------------------------------------------------------
__global__ void cvt_f16_k(const float* __restrict__ in, half_t* __restrict__ out, long long n) {
  long long i = (long long)blockIdx.x * 256 + threadIdx.x;
  if (i < n) out[i] = (half_t)in[i];
}

// ---------------------------------------------------------------------------
// rmsnorm over last dim (=1024); writes optional f32 and mandatory f16
// ---------------------------------------------------------------------------
__global__ __launch_bounds__(256) void rmsnorm_k(const float* __restrict__ x,
                                                 const float* __restrict__ w,
                                                 float* __restrict__ of32,
                                                 half_t* __restrict__ of16) {
  const int t = blockIdx.x;
  const int tid = threadIdx.x;
  const float* row = x + (long long)t * DIM_;
  float ss = 0.f;
  for (int i = tid; i < DIM_; i += 256) { float v = row[i]; ss += v * v; }
  __shared__ float red[256];
  red[tid] = ss; __syncthreads();
  for (int off = 128; off > 0; off >>= 1) {
    if (tid < off) red[tid] += red[tid + off];
    __syncthreads();
  }
  float r = rsqrtf(red[0] * (1.f / DIM_) + 1e-6f);
  for (int i = tid; i < DIM_; i += 256) {
    float v = row[i] * r * w[i];
    if (of32) of32[(long long)t * DIM_ + i] = v;
    of16[(long long)t * DIM_ + i] = (half_t)v;
  }
}

// ---------------------------------------------------------------------------
// Generic WMMA GEMM:  C[M,N] = A[M,K] @ B[K,N] (+ bias) (+ residual)
//   - A,B f16 row-major in global; B staged TRANSPOSED in LDS so both WMMA
//     fragments are contiguous, aligned ds_load_b128 reads.
//   - optional expert grouping (groupOff), A row-gather (arow), C row-scatter
//     (crow).  Block: 128 threads = 4 waves; tile 64x64; K-step 32.
// ---------------------------------------------------------------------------
#define BLK_M 64
#define BLK_N 64
#define BLK_K 32

__global__ __launch_bounds__(128) void gemm_wmma(
    const half_t* __restrict__ A, int lda,
    const half_t* __restrict__ Ball, int ldb, long long bstride,
    const float* __restrict__ biasAll, int biasStride,
    const float* __restrict__ residual, int ldr,
    float* __restrict__ C, int ldc,
    int M, int N, int K,
    const int* __restrict__ arow,
    const int* __restrict__ crow,
    const int* __restrict__ groupOff) {
  __shared__ __attribute__((aligned(32))) half_t As[BLK_M][40];  // 80B rows (16B-mult)
  __shared__ __attribute__((aligned(32))) half_t Bt[BLK_N][48];  // 96B rows (32B-mult)

  const int e = groupOff ? blockIdx.z : 0;
  int mBegin, mEnd;
  if (groupOff) { mBegin = groupOff[e]; mEnd = groupOff[e + 1]; }
  else          { mBegin = 0;           mEnd = M; }
  const int m0 = mBegin + blockIdx.y * BLK_M;
  if (m0 >= mEnd) return;
  const int mValid = mEnd - m0;
  const int n0 = blockIdx.x * BLK_N;

  const half_t* B = Ball + (long long)e * bstride;
  const float* bias = biasAll ? biasAll + (long long)e * biasStride : (const float*)0;

  const int tid  = threadIdx.x;
  const int wave = tid >> 5;
  const int lane = tid & 31;
  const int hi   = lane >> 4;
  const int l15  = lane & 15;

  v8f acc[4];
  v8f vzero = {};
#pragma unroll
  for (int i = 0; i < 4; ++i) acc[i] = vzero;

  for (int k0 = 0; k0 < K; k0 += BLK_K) {
    if (k0 + BLK_K < K)  // prefetch next B tile (global_prefetch_b8)
      __builtin_prefetch((const void*)(B + (long long)(k0 + BLK_K) * ldb + n0), 0, 1);
    // A tile: 64x32, 2 threads/row, 2x b128 load + 2x b128 LDS store
    {
      int r = tid >> 1;
      int cb = (tid & 1) * 16;
      int mr = (r < mValid) ? (m0 + r) : m0;
      int ga = arow ? arow[mr] : mr;
      const v8h* src = (const v8h*)(A + (long long)ga * lda + k0 + cb);
      *(v8h*)&As[r][cb]     = src[0];
      *(v8h*)&As[r][cb + 8] = src[1];
    }
    // B tile: 32x64 row-major global (coalesced b128 loads),
    // scattered b16 stores into transposed LDS tile Bt[col][k]
    {
      int r  = tid >> 2;          // k row 0..31
      int cb = (tid & 3) * 16;    // col base
      const v8h* src = (const v8h*)(B + (long long)(k0 + r) * ldb + n0 + cb);
      v8h g0 = src[0], g1 = src[1];
#pragma unroll
      for (int c = 0; c < 8; ++c) {
        Bt[cb + c][r]     = g0[c];
        Bt[cb + 8 + c][r] = g1[c];
      }
    }
    __syncthreads();

    // A fragment (16x32): lane row = l15, K pairs split by lane half
    v16h af;
    {
      int r = wave * 16 + l15;
      int kb = hi * 16;
      v8h a0 = *(const v8h*)&As[r][kb];
      v8h a1 = *(const v8h*)&As[r][kb + 8];
#pragma unroll
      for (int j = 0; j < 8; ++j) { af[j] = a0[j]; af[j + 8] = a1[j]; }
    }
#pragma unroll
    for (int nt = 0; nt < 4; ++nt) {
      // B fragment (32x16): lane col = l15, K = hi*16 + j  -> contiguous in Bt
      int ccol = nt * 16 + l15;
      v16h bf = *(const v16h*)&Bt[ccol][hi * 16];
      acc[nt] = __builtin_amdgcn_wmma_f32_16x16x32_f16(
          false, af, false, bf, (short)0, acc[nt], false, false);
    }
    __syncthreads();
  }

  // epilogue: D layout -> lane col = l15, row = wave*16 + hi*8 + i
#pragma unroll
  for (int nt = 0; nt < 4; ++nt) {
    int gcol = n0 + nt * 16 + l15;
#pragma unroll
    for (int i = 0; i < 8; ++i) {
      int rloc = wave * 16 + hi * 8 + i;
      if (rloc >= mValid) continue;
      int m = m0 + rloc;
      int cr = crow ? crow[m] : m;
      float v = acc[nt][i];
      if (bias) v += bias[gcol];
      if (residual) v += residual[(long long)cr * ldr + gcol];
      C[(long long)cr * ldc + gcol] = v;
    }
  }
}

// ---------------------------------------------------------------------------
// RoPE + per-head rmsnorm for Q (16 heads) and K (4 heads)
// qkv layout: [S, 1536] = [q(1024) | k(256) | v(256)]
// ---------------------------------------------------------------------------
__global__ __launch_bounds__(64) void rope_rms_qk(const float* __restrict__ qkv,
                                                  const float* __restrict__ qnw,
                                                  const float* __restrict__ knw,
                                                  half_t* __restrict__ Qo,
                                                  half_t* __restrict__ Ko) {
  const int t  = blockIdx.x;
  const int hh = blockIdx.y;
  const int d  = threadIdx.x;
  const bool isQ = hh < NQ_;
  const int h = isQ ? hh : hh - NQ_;
  const float* src = qkv + (long long)t * 1536 + (isQ ? h * HD_ : DIM_ + h * HD_);

  int j = d & 31;
  float inv_freq = __expf(-(float)j * 0.03125f * 9.210340371976184f); // 10000^{-2j/64}
  float ang = (float)t * inv_freq;
  float c = __cosf(ang), s = __sinf(ang);
  float x1 = src[j], x2 = src[j + 32];
  float val = (d < 32) ? (x1 * c - x2 * s) : (x1 * s + x2 * c);

  __shared__ float red[64];
  red[d] = val * val;
  __syncthreads();
  for (int off = 32; off > 0; off >>= 1) {
    if (d < off) red[d] += red[d + off];
    __syncthreads();
  }
  float r = rsqrtf(red[0] * (1.f / HD_) + 1e-6f);
  float w = isQ ? qnw[d] : knw[d];
  float o = w * val * r;
  if (isQ) Qo[((long long)t * NQ_ + h) * HD_ + d] = (half_t)o;
  else     Ko[((long long)t * NKV_ + h) * HD_ + d] = (half_t)o;
}

__global__ void vslice_k(const float* __restrict__ qkv, half_t* __restrict__ v) {
  int i = blockIdx.x * 256 + threadIdx.x;  // S * 256
  if (i < S_ * NKV_ * HD_) {
    int t = i >> 8, c = i & 255;
    v[i] = (half_t)qkv[(long long)t * 1536 + 1280 + c];
  }
}

// ---------------------------------------------------------------------------
// Single-wave flash attention with WMMA.  grid = (S/16, NQ)
//   Ks rows padded to 80 halves (160B, 32B-mult) so the QK^T B-fragment is
//   one aligned v16h LDS read; V staged transposed (VsT[d][key]) so the PV
//   B-fragment is likewise a single aligned v16h read.
// ---------------------------------------------------------------------------
__global__ __launch_bounds__(32) void flash_attn(const half_t* __restrict__ Q,
                                                 const half_t* __restrict__ Kg,
                                                 const half_t* __restrict__ Vg,
                                                 float* __restrict__ Of32,
                                                 half_t* __restrict__ Of16) {
  __shared__ __attribute__((aligned(32))) half_t Qs[16][80];
  __shared__ __attribute__((aligned(32))) half_t Ks[32][80];
  __shared__ __attribute__((aligned(32))) half_t VsT[64][48];
  __shared__ __attribute__((aligned(32))) half_t Ps[16][40];

  const int qt0 = blockIdx.x * 16;
  const int h   = blockIdx.y;
  const int kvh = h >> 2;  // rep = NQ/NKV = 4

  const int lane = threadIdx.x;
  const int hi = lane >> 4, l15 = lane & 15;

  {  // load Q tile 16x64 (4x b128 per lane)
    const v8h* src = (const v8h*)(Q + ((long long)(qt0 + l15) * NQ_ + h) * HD_ + hi * 32);
#pragma unroll
    for (int q8 = 0; q8 < 4; ++q8) *(v8h*)&Qs[l15][hi * 32 + q8 * 8] = src[q8];
  }
  __syncthreads();

  v16h aq[2];
#pragma unroll
  for (int c = 0; c < 2; ++c) {
    int kb = c * 32 + hi * 16;
    v8h a0 = *(const v8h*)&Qs[l15][kb];
    v8h a1 = *(const v8h*)&Qs[l15][kb + 8];
#pragma unroll
    for (int j = 0; j < 8; ++j) { aq[c][j] = a0[j]; aq[c][j + 8] = a1[j]; }
  }

  v8f accO[4];
  v8f vzero = {};
#pragma unroll
  for (int i = 0; i < 4; ++i) accO[i] = vzero;
  float mrow[8], lrow[8];
#pragma unroll
  for (int i = 0; i < 8; ++i) { mrow[i] = -1e30f; lrow[i] = 0.f; }

  const int nkb = (qt0 + 16 + 31) / 32;
  for (int kbI = 0; kbI < nkb; ++kbI) {
    const int kb0 = kbI * 32;
    {  // K row-major; V transposed into LDS (b16 scatter stores)
      const v8h* ksrc = (const v8h*)(Kg + ((long long)(kb0 + lane) * NKV_ + kvh) * HD_);
      const v8h* vsrc = (const v8h*)(Vg + ((long long)(kb0 + lane) * NKV_ + kvh) * HD_);
#pragma unroll
      for (int c8 = 0; c8 < 8; ++c8) {
        *(v8h*)&Ks[lane][c8 * 8] = ksrc[c8];
        v8h vv = vsrc[c8];
#pragma unroll
        for (int c = 0; c < 8; ++c) VsT[c8 * 8 + c][lane] = vv[c];
      }
    }
    __syncthreads();

    // scores = Q (16x64) @ K^T (64x32) -> two 16x16 tiles
    v8f sc[2];
#pragma unroll
    for (int nt = 0; nt < 2; ++nt) {
      sc[nt] = vzero;
#pragma unroll
      for (int c = 0; c < 2; ++c) {
        int key = nt * 16 + l15;
        v16h bf = *(const v16h*)&Ks[key][c * 32 + hi * 16];
        sc[nt] = __builtin_amdgcn_wmma_f32_16x16x32_f16(
            false, aq[c], false, bf, (short)0, sc[nt], false, false);
      }
    }

    // scale + causal mask
#pragma unroll
    for (int nt = 0; nt < 2; ++nt) {
      int key = kb0 + nt * 16 + l15;
#pragma unroll
      for (int i = 0; i < 8; ++i) {
        int qrow = qt0 + hi * 8 + i;
        float v = sc[nt][i] * 0.125f;
        if (key > qrow) v = -1e30f;
        sc[nt][i] = v;
      }
    }

    // online softmax (row reductions across 16-lane half-wave)
    float alpha[8];
#pragma unroll
    for (int i = 0; i < 8; ++i) {
      float mx = fmaxf(sc[0][i], sc[1][i]);
      for (int off = 1; off < 16; off <<= 1) mx = fmaxf(mx, __shfl_xor(mx, off, 32));
      float mnew = fmaxf(mrow[i], mx);
      alpha[i] = __expf(mrow[i] - mnew);
      mrow[i] = mnew;
      float p0 = __expf(sc[0][i] - mnew);
      float p1 = __expf(sc[1][i] - mnew);
      sc[0][i] = p0; sc[1][i] = p1;
      float ls = p0 + p1;
      for (int off = 1; off < 16; off <<= 1) ls += __shfl_xor(ls, off, 32);
      lrow[i] = lrow[i] * alpha[i] + ls;
    }
#pragma unroll
    for (int nt = 0; nt < 4; ++nt)
#pragma unroll
      for (int i = 0; i < 8; ++i) accO[nt][i] *= alpha[i];

    __syncthreads();
#pragma unroll
    for (int i = 0; i < 8; ++i) {
      Ps[hi * 8 + i][l15]      = (half_t)sc[0][i];
      Ps[hi * 8 + i][16 + l15] = (half_t)sc[1][i];
    }
    __syncthreads();

    // O += P (16x32) @ V (32x64)
    v16h ap;
    {
      int kb = hi * 16;
      v8h a0 = *(const v8h*)&Ps[l15][kb];
      v8h a1 = *(const v8h*)&Ps[l15][kb + 8];
#pragma unroll
      for (int j = 0; j < 8; ++j) { ap[j] = a0[j]; ap[j + 8] = a1[j]; }
    }
#pragma unroll
    for (int nt = 0; nt < 4; ++nt) {
      int dcol = nt * 16 + l15;
      v16h bf = *(const v16h*)&VsT[dcol][hi * 16];
      accO[nt] = __builtin_amdgcn_wmma_f32_16x16x32_f16(
          false, ap, false, bf, (short)0, accO[nt], false, false);
    }
    __syncthreads();
  }

#pragma unroll
  for (int nt = 0; nt < 4; ++nt)
#pragma unroll
    for (int i = 0; i < 8; ++i) {
      int qrow = qt0 + hi * 8 + i;
      float v = accO[nt][i] / lrow[i];
      int col = h * HD_ + nt * 16 + l15;
      Of32[(long long)qrow * DIM_ + col] = v;
      Of16[(long long)qrow * DIM_ + col] = (half_t)v;
    }
}

// ---------------------------------------------------------------------------
// gate multiply, router, MoE plumbing, swiglu
// ---------------------------------------------------------------------------
__global__ void gate_mul_k(const float* __restrict__ attn, const float* __restrict__ gate,
                           half_t* __restrict__ out, long long n) {
  long long i = (long long)blockIdx.x * 256 + threadIdx.x;
  if (i < n) out[i] = (half_t)(attn[i] * (1.f / (1.f + __expf(-gate[i]))));
}

__global__ __launch_bounds__(64) void router_k(const float* __restrict__ h2,
                                               const float* __restrict__ Wr,
                                               const float* __restrict__ br,
                                               int* __restrict__ top1,
                                               int* __restrict__ counts) {
  const int t = blockIdx.x, tid = threadIdx.x;
  float acc[E_];
#pragma unroll
  for (int e = 0; e < E_; ++e) acc[e] = 0.f;
  const float* row = h2 + (long long)t * DIM_;
  for (int d = tid; d < DIM_; d += 64) {
    float xv = row[d];
    const float* wr = Wr + d * E_;
#pragma unroll
    for (int e = 0; e < E_; ++e) acc[e] += xv * wr[e];
  }
  __shared__ float red[64][E_];
#pragma unroll
  for (int e = 0; e < E_; ++e) red[tid][e] = acc[e];
  __syncthreads();
  for (int off = 32; off > 0; off >>= 1) {
    if (tid < off)
      for (int e = 0; e < E_; ++e) red[tid][e] += red[tid + off][e];
    __syncthreads();
  }
  if (tid == 0) {
    int best = 0; float bv = red[0][0] + br[0];
    for (int e = 1; e < E_; ++e) {
      float v = red[0][e] + br[e];
      if (v > bv) { bv = v; best = e; }
    }
    top1[t] = best;
    atomicAdd(&counts[best], 1);
  }
}

__global__ void init_counts_k(int* counts, int* cursor) {
  int i = threadIdx.x;
  if (i < E_) { counts[i] = 0; cursor[i] = 0; }
}

__global__ void offsets_k(const int* counts, int* offsets) {
  if (threadIdx.x == 0) {
    int s = 0;
    for (int e = 0; e < E_; ++e) { offsets[e] = s; s += counts[e]; }
    offsets[E_] = s;
  }
}

__global__ void assign_k(const int* top1, const int* offsets, int* cursor, int* perm) {
  int t = blockIdx.x * 256 + threadIdx.x;
  if (t < S_) {
    int e = top1[t];
    int pos = atomicAdd(&cursor[e], 1);
    perm[offsets[e] + pos] = t;
  }
}

__global__ void swiglu_k(const float* __restrict__ u, half_t* __restrict__ act) {
  long long i = (long long)blockIdx.x * 256 + threadIdx.x;  // S * HID
  if (i < (long long)S_ * HID_) {
    long long t = i >> 11, j = i & (HID_ - 1);
    float a = u[t * (2 * HID_) + j];
    float b = u[t * (2 * HID_) + HID_ + j];
    act[i] = (half_t)(a * (1.f / (1.f + __expf(-b))));
  }
}

// ---------------------------------------------------------------------------
extern "C" void kernel_launch(void* const* d_in, const int* in_sizes, int n_in,
                              void* d_out, int out_size, void* d_ws, size_t ws_size,
                              hipStream_t stream) {
  const float* x    = (const float*)d_in[0];
  const float* ln1w = (const float*)d_in[1];
  const float* ln2w = (const float*)d_in[2];
  const float* Wq   = (const float*)d_in[3];
  const float* bq   = (const float*)d_in[4];
  const float* Wk   = (const float*)d_in[5];
  const float* bk   = (const float*)d_in[6];
  const float* Wv   = (const float*)d_in[7];
  const float* bv   = (const float*)d_in[8];
  const float* qnw  = (const float*)d_in[9];
  const float* knw  = (const float*)d_in[10];
  const float* Wg   = (const float*)d_in[11];
  const float* bg   = (const float*)d_in[12];
  const float* Wo   = (const float*)d_in[13];
  const float* bo   = (const float*)d_in[14];
  const float* Wr   = (const float*)d_in[15];
  const float* br   = (const float*)d_in[16];
  const float* W1   = (const float*)d_in[17];
  const float* b1   = (const float*)d_in[18];
  const float* W2   = (const float*)d_in[19];
  const float* b2   = (const float*)d_in[20];
  float* out = (float*)d_out;
  (void)in_sizes; (void)n_in; (void)out_size; (void)ws_size;

  size_t off = 0;
  auto alloc = [&](size_t bytes) -> char* {
    char* p = (char*)d_ws + off;
    off = (off + bytes + 255) & ~(size_t)255;
    return p;
  };
  half_t* h1f16  = (half_t*)alloc((size_t)S_ * DIM_ * 2);
  float*  qkv    = (float*) alloc((size_t)S_ * 1536 * 4);
  half_t* qf16   = (half_t*)alloc((size_t)S_ * NQ_ * HD_ * 2);
  half_t* kf16   = (half_t*)alloc((size_t)S_ * NKV_ * HD_ * 2);
  half_t* vf16   = (half_t*)alloc((size_t)S_ * NKV_ * HD_ * 2);
  float*  attn32 = (float*) alloc((size_t)S_ * DIM_ * 4);
  half_t* attn16 = (half_t*)alloc((size_t)S_ * DIM_ * 2);
  float*  gate32 = (float*) alloc((size_t)S_ * DIM_ * 4);
  half_t* gated  = (half_t*)alloc((size_t)S_ * DIM_ * 2);
  float*  x2     = (float*) alloc((size_t)S_ * DIM_ * 4);
  float*  h2f32  = (float*) alloc((size_t)S_ * DIM_ * 4);
  half_t* h2f16  = (half_t*)alloc((size_t)S_ * DIM_ * 2);
  float*  ubuf   = (float*) alloc((size_t)S_ * 2 * HID_ * 4);
  half_t* act16  = (half_t*)alloc((size_t)S_ * HID_ * 2);
  half_t* wq16   = (half_t*)alloc((size_t)DIM_ * DIM_ * 2);
  half_t* wk16   = (half_t*)alloc((size_t)DIM_ * NKV_ * HD_ * 2);
  half_t* wv16   = (half_t*)alloc((size_t)DIM_ * NKV_ * HD_ * 2);
  half_t* wg16   = (half_t*)alloc((size_t)HD_ * HD_ * 2);
  half_t* wo16   = (half_t*)alloc((size_t)DIM_ * DIM_ * 2);
  half_t* w116   = (half_t*)alloc((size_t)E_ * DIM_ * 2 * HID_ * 2);
  half_t* w216   = (half_t*)alloc((size_t)E_ * HID_ * DIM_ * 2);
  int* top1    = (int*)alloc(S_ * 4);
  int* counts  = (int*)alloc(E_ * 4);
  int* offsets = (int*)alloc((E_ + 1) * 4);
  int* cursor  = (int*)alloc(E_ * 4);
  int* perm    = (int*)alloc(S_ * 4);

  auto cvt = [&](const float* src, half_t* dst, long long n) {
    cvt_f16_k<<<(unsigned)((n + 255) / 256), 256, 0, stream>>>(src, dst, n);
  };
  cvt(Wq, wq16, (long long)DIM_ * DIM_);
  cvt(Wk, wk16, (long long)DIM_ * NKV_ * HD_);
  cvt(Wv, wv16, (long long)DIM_ * NKV_ * HD_);
  cvt(Wg, wg16, (long long)HD_ * HD_);
  cvt(Wo, wo16, (long long)DIM_ * DIM_);
  cvt(W1, w116, (long long)E_ * DIM_ * 2 * HID_);
  cvt(W2, w216, (long long)E_ * HID_ * DIM_);

  // h1 = rmsnorm(x, ln1)
  rmsnorm_k<<<S_, 256, 0, stream>>>(x, ln1w, (float*)0, h1f16);

  // QKV GEMMs into fused [S,1536] buffer
  gemm_wmma<<<dim3(DIM_ / 64, S_ / 64, 1), 128, 0, stream>>>(
      h1f16, DIM_, wq16, DIM_, 0, bq, 0, (const float*)0, 0,
      qkv, 1536, S_, DIM_, DIM_, (const int*)0, (const int*)0, (const int*)0);
  gemm_wmma<<<dim3(256 / 64, S_ / 64, 1), 128, 0, stream>>>(
      h1f16, DIM_, wk16, 256, 0, bk, 0, (const float*)0, 0,
      qkv + 1024, 1536, S_, 256, DIM_, (const int*)0, (const int*)0, (const int*)0);
  gemm_wmma<<<dim3(256 / 64, S_ / 64, 1), 128, 0, stream>>>(
      h1f16, DIM_, wv16, 256, 0, bv, 0, (const float*)0, 0,
      qkv + 1280, 1536, S_, 256, DIM_, (const int*)0, (const int*)0, (const int*)0);

  // RoPE + head rmsnorm; V slice convert
  rope_rms_qk<<<dim3(S_, NQ_ + NKV_, 1), 64, 0, stream>>>(qkv, qnw, knw, qf16, kf16);
  vslice_k<<<(S_ * NKV_ * HD_) / 256, 256, 0, stream>>>(qkv, vf16);

  // flash attention
  flash_attn<<<dim3(S_ / 16, NQ_, 1), 32, 0, stream>>>(qf16, kf16, vf16, attn32, attn16);

  // gate = attn @ Wg + bg  (M = S*NQ, K = N = 64)
  gemm_wmma<<<dim3(1, (S_ * NQ_) / 64, 1), 128, 0, stream>>>(
      attn16, HD_, wg16, HD_, 0, bg, 0, (const float*)0, 0,
      gate32, HD_, S_ * NQ_, HD_, HD_, (const int*)0, (const int*)0, (const int*)0);
  gate_mul_k<<<(S_ * DIM_) / 256, 256, 0, stream>>>(attn32, gate32, gated,
                                                    (long long)S_ * DIM_);

  // x2 = x + gated @ Wo + bo
  gemm_wmma<<<dim3(DIM_ / 64, S_ / 64, 1), 128, 0, stream>>>(
      gated, DIM_, wo16, DIM_, 0, bo, 0, x, DIM_,
      x2, DIM_, S_, DIM_, DIM_, (const int*)0, (const int*)0, (const int*)0);

  // h2 = rmsnorm(x2, ln2)
  rmsnorm_k<<<S_, 256, 0, stream>>>(x2, ln2w, h2f32, h2f16);

  // router + top1, build per-expert token lists
  init_counts_k<<<1, 32, 0, stream>>>(counts, cursor);
  router_k<<<S_, 64, 0, stream>>>(h2f32, Wr, br, top1, counts);
  offsets_k<<<1, 32, 0, stream>>>(counts, offsets);
  assign_k<<<(S_ + 255) / 256, 256, 0, stream>>>(top1, offsets, cursor, perm);

  // MoE grouped GEMM 1: u[slot] = h2[perm[slot]] @ W1[e] + b1[e]
  gemm_wmma<<<dim3((2 * HID_) / 64, S_ / 64, E_), 128, 0, stream>>>(
      h2f16, DIM_, w116, 2 * HID_, (long long)DIM_ * 2 * HID_, b1, 2 * HID_,
      (const float*)0, 0, ubuf, 2 * HID_, S_, 2 * HID_, DIM_,
      perm, (const int*)0, offsets);
  swiglu_k<<<((long long)S_ * HID_) / 256, 256, 0, stream>>>(ubuf, act16);

  // MoE grouped GEMM 2: out[perm[slot]] = x2[perm[slot]] + act[slot] @ W2[e] + b2[e]
  gemm_wmma<<<dim3(DIM_ / 64, S_ / 64, E_), 128, 0, stream>>>(
      act16, HID_, w216, DIM_, (long long)HID_ * DIM_, b2, DIM_,
      x2, DIM_, out, DIM_, S_, DIM_, HID_,
      (const int*)0, perm, offsets);
}
